// BCQLinear_88132728914380
// MI455X (gfx1250) — compile-verified, matrix operand order checked by reading, not run on hardware
//
#include <hip/hip_runtime.h>
#include <hip/hip_bf16.h>
#include <stdint.h>

// Problem dims (match reference)
#define IN_F      4096
#define OUT_F     4096
#define GROUP     128
#define INTERVAL  32
#define NBITS_MAX 4
#define BATCH     8192
#define JB_CNT    (OUT_F / INTERVAL)   // 128

typedef __attribute__((ext_vector_type(16))) __bf16 v16bf;
typedef __attribute__((ext_vector_type(8)))  float  v8f;

// Hardware packed f32->bf16 convert (RNE), one VALU op for two values.
static __device__ __forceinline__ unsigned pk_bf16(float lo, float hi) {
  unsigned r;
  asm("v_cvt_pk_bf16_f32 %0, %1, %2" : "=v"(r) : "v"(lo), "v"(hi));
  return r;
}
static __device__ __forceinline__ unsigned short f32_to_bf16_rne(float f) {
  return (unsigned short)(pk_bf16(f, 0.0f) & 0xFFFFu);
}

// CDNA5 async global->LDS copy (ASYNCcnt-tracked, bypasses VGPRs).
// lds_addr = wave-relative LDS byte address (low 32 bits of generic pointer);
// gaddr    = per-lane 64-bit global address. Copies 2x16B (offset 0 and 16
// applied to BOTH the LDS and global side per the ISA addressing rule).
static __device__ __forceinline__ void async_copy_b128x2(unsigned lds_addr,
                                                         const void* gaddr) {
  asm volatile(
      "global_load_async_to_lds_b128 %0, %1, off\n\t"
      "global_load_async_to_lds_b128 %0, %1, off offset:16"
      :: "v"(lds_addr), "v"(gaddr) : "memory");
}
static __device__ __forceinline__ void wait_async0() {
  asm volatile("s_wait_asynccnt 0x0" ::: "memory");
}

// ---------------------------------------------------------------------------
// Kernel 1: dequantize BCQ weights and scatter through BOTH permutations.
//   WT[out_reorder[n]][in_reorder[k]] = alpha[g,n] * (2*val - (2^nb - 1)) + beta[g,n]
// One thread per (kw, n): kw covers 32 consecutive k's (one packed uint32 strip).
// ---------------------------------------------------------------------------
__global__ __launch_bounds__(256) void bcq_dequant_scatter(
    const unsigned int* __restrict__ qweight,        // [IN_F/32, 4, OUT_F]
    const float*        __restrict__ alpha,          // [G, OUT_F]
    const float*        __restrict__ beta,           // [G, OUT_F]
    const int*          __restrict__ in_reorder,     // [IN_F]
    const int*          __restrict__ out_reorder,    // [OUT_F]
    const int*          __restrict__ block_bitwidth, // [G, JB]
    unsigned short*     __restrict__ wT)             // out: [OUT_F][IN_F] bf16
{
  const int n  = blockIdx.x * 256 + threadIdx.x;  // 0..OUT_F-1
  const int kw = blockIdx.y;                      // 0..IN_F/32-1
  const int g  = kw >> 2;                         // 32 k's stay in one group of 128
  const int jb = n >> 5;

  const int   nb = block_bitwidth[g * JB_CNT + jb];
  const float a  = alpha[g * OUT_F + n];
  const float be = beta [g * OUT_F + n];
  const unsigned w0 = qweight[(kw * 4 + 0) * OUT_F + n];
  const unsigned w1 = qweight[(kw * 4 + 1) * OUT_F + n];
  const unsigned w2 = qweight[(kw * 4 + 2) * OUT_F + n];
  const unsigned w3 = qweight[(kw * 4 + 3) * OUT_F + n];
  const unsigned msk = (1u << nb) - 1u;
  const float   offs = (float)(int)msk;           // 2^nb - 1

  unsigned short* row = wT + (size_t)out_reorder[n] * IN_F;
  const int kbase = kw * 32;
  #pragma unroll 8
  for (int i = 0; i < 32; ++i) {
    unsigned v = ((w0 >> i) & 1u)
               | (((w1 >> i) & 1u) << 1)
               | (((w2 >> i) & 1u) << 2)
               | (((w3 >> i) & 1u) << 3);
    v &= msk;
    float wv = a * (2.0f * (float)v - offs) + be;
    row[in_reorder[kbase + i]] = f32_to_bf16_rne(wv);   // uniform index per warp-iter
  }
}

// ---------------------------------------------------------------------------
// Kernel 2: out[BATCH][OUT_F] = x(f32->bf16) @ WT^T via v_wmma_f32_16x16x32_bf16
// 256 threads = 8 waves; block tile 128(M) x 128(N); K-step 32; double-buffered.
// B tile: global_load_async_to_lds_b128 (ASYNCcnt). A tile: f32->bf16 via VGPRs.
// Wave (wm,wn) in 2x4 grid owns a 64x32 subtile = 4x2 WMMA accumulators.
// ---------------------------------------------------------------------------
#define BM 128
#define BN 128
#define BK 32
#define NT (IN_F / BK)  // 128 K-steps
#define LDSS 40         // padded LDS row stride (80 B: 16B-aligned, conflict-free)
#define ABUF (BM * LDSS)
#define BBUF (BN * LDSS)

union Frag { v16bf v; uint4 q[2]; };

__global__ __launch_bounds__(256) void bcq_gemm_wmma(
    const float*          __restrict__ x,    // [BATCH][IN_F] f32
    const unsigned short* __restrict__ wT,   // [OUT_F][IN_F] bf16
    float*                __restrict__ out)  // [BATCH][OUT_F] f32
{
  __shared__ __align__(16) unsigned short sA[2 * ABUF];
  __shared__ __align__(16) unsigned short sB[2 * BBUF];

  const int tid  = threadIdx.x;
  const int wave = tid >> 5;
  const int lane = tid & 31;
  const int kh   = lane >> 4;     // lane-half selects K sub-chunk (ISA 16-bit layout)
  const int nl   = lane & 15;     // row within fragment
  const int wm   = (wave >> 2) * 64;   // wave M offset in block tile
  const int wn   = (wave & 3) * 32;    // wave N offset in block tile
  const int bm   = blockIdx.x * BM;
  const int bn   = blockIdx.y * BN;

  // cooperative loader mapping
  const int arow = tid >> 1;            // 0..127 (M row)
  const int acol = (tid & 1) << 4;      // 0 or 16 (K offset, 16 f32)
  const int brow = tid & 127;           // 0..127 (N row of wT)
  const int bcol = (tid >> 7) << 4;     // 0 or 16 (K offset, 16 bf16)
  const int aoff = arow * LDSS + acol;  // A staging offset within a buffer

  const float*          xbase = x  + (size_t)(bm + arow) * IN_F + acol;
  const unsigned short* wbase = wT + (size_t)(bn + brow) * IN_F + bcol;

  // 32-bit LDS byte addresses for the async B copy targets
  const unsigned sBlds0 = (unsigned)(uintptr_t)&sB[brow * LDSS + bcol];
  const unsigned sBlds1 = sBlds0 + (unsigned)(BBUF * sizeof(unsigned short));

  v8f acc[4][2];
  #pragma unroll
  for (int i = 0; i < 4; ++i)
    #pragma unroll
    for (int j = 0; j < 2; ++j)
      acc[i][j] = (v8f){};

  // ---- prologue: stage tile 0 into buffer 0 ----
  {
    async_copy_b128x2(sBlds0, wbase);
    float4 a0 = *(const float4*)(xbase + 0);
    float4 a1 = *(const float4*)(xbase + 4);
    float4 a2 = *(const float4*)(xbase + 8);
    float4 a3 = *(const float4*)(xbase + 12);
    uint4 pa0, pa1;
    pa0.x = pk_bf16(a0.x, a0.y); pa0.y = pk_bf16(a0.z, a0.w);
    pa0.z = pk_bf16(a1.x, a1.y); pa0.w = pk_bf16(a1.z, a1.w);
    pa1.x = pk_bf16(a2.x, a2.y); pa1.y = pk_bf16(a2.z, a2.w);
    pa1.z = pk_bf16(a3.x, a3.y); pa1.w = pk_bf16(a3.z, a3.w);
    *(uint4*)(&sA[aoff])     = pa0;    // direct GEP on shared array -> ds_store
    *(uint4*)(&sA[aoff + 8]) = pa1;
    wait_async0();
    __syncthreads();
  }

  int koff = BK;  // next tile's K offset (wraps to 0 for the final dummy prefetch)
  for (int t = 0; t < NT; ++t) {
    const int cur = (t & 1) ? BBUF : 0;        // B buffer base (elements)
    const int curA = (t & 1) ? ABUF : 0;       // A buffer base (elements)
    const int nxtA = curA ^ ABUF;

    // 1) issue next-tile traffic: B straight to LDS (async), A f32 to regs
    async_copy_b128x2((t & 1) ? sBlds0 : sBlds1, wbase + koff);
    float4 a0 = *(const float4*)(xbase + koff + 0);
    float4 a1 = *(const float4*)(xbase + koff + 4);
    float4 a2 = *(const float4*)(xbase + koff + 8);
    float4 a3 = *(const float4*)(xbase + koff + 12);
    koff += BK;
    if (koff == IN_F) koff = 0;   // uniform wrap: last prefetch is a harmless dummy

    // 2) compute current tile
    // fragment loads: lane<16 -> K{0..7,16..23}; lane>=16 -> K{8..15,24..31}
    Frag A[4], B[2];
    #pragma unroll
    for (int i = 0; i < 4; ++i) {
      const unsigned short* p = &sA[curA + (wm + i * 16 + nl) * LDSS + kh * 8];
      A[i].q[0] = *(const uint4*)(p);
      A[i].q[1] = *(const uint4*)(p + 16);
    }
    #pragma unroll
    for (int j = 0; j < 2; ++j) {
      const unsigned short* p = &sB[cur + (wn + j * 16 + nl) * LDSS + kh * 8];
      B[j].q[0] = *(const uint4*)(p);
      B[j].q[1] = *(const uint4*)(p + 16);
    }
    #pragma unroll
    for (int i = 0; i < 4; ++i)
      #pragma unroll
      for (int j = 0; j < 2; ++j)
        acc[i][j] = __builtin_amdgcn_wmma_f32_16x16x32_bf16(
            false, A[i].v, false, B[j].v, (short)0, acc[i][j], false, false);

    // 3) stage next A tile into LDS (convert after compute to hide load latency)
    uint4 pa0, pa1;
    pa0.x = pk_bf16(a0.x, a0.y); pa0.y = pk_bf16(a0.z, a0.w);
    pa0.z = pk_bf16(a1.x, a1.y); pa0.w = pk_bf16(a1.z, a1.w);
    pa1.x = pk_bf16(a2.x, a2.y); pa1.y = pk_bf16(a2.z, a2.w);
    pa1.z = pk_bf16(a3.x, a3.y); pa1.w = pk_bf16(a3.z, a3.w);
    *(uint4*)(&sA[nxtA + aoff])     = pa0;   // direct GEP -> ds_store_b128
    *(uint4*)(&sA[nxtA + aoff + 8]) = pa1;

    // 4) publish: own async B done, then workgroup barrier (barrier waits dscnt)
    wait_async0();
    __syncthreads();
  }

  // epilogue: C/D layout — VGPR r, lanes 0-15: M=r, N=lane; lanes 16-31: M=r+8
  #pragma unroll
  for (int i = 0; i < 4; ++i) {
    #pragma unroll
    for (int j = 0; j < 2; ++j) {
      #pragma unroll
      for (int r = 0; r < 8; ++r) {
        int m = bm + wm + i * 16 + kh * 8 + r;
        int n = bn + wn + j * 16 + nl;
        out[(size_t)m * OUT_F + n] = acc[i][j][r];
      }
    }
  }
}

// ---------------------------------------------------------------------------
extern "C" void kernel_launch(void* const* d_in, const int* in_sizes, int n_in,
                              void* d_out, int out_size, void* d_ws, size_t ws_size,
                              hipStream_t stream) {
  const float*        x           = (const float*)d_in[0];
  const unsigned int* qweight     = (const unsigned int*)d_in[1];
  const float*        alpha       = (const float*)d_in[2];
  const float*        beta        = (const float*)d_in[3];
  const int*          in_reorder  = (const int*)d_in[4];
  const int*          out_reorder = (const int*)d_in[5];
  const int*          block_bw    = (const int*)d_in[6];
  float*              out         = (float*)d_out;

  // workspace: permuted bf16 weight, [OUT_F][IN_F] = 32 MB
  unsigned short* wT = (unsigned short*)d_ws;

  bcq_dequant_scatter<<<dim3(OUT_F / 256, IN_F / 32), 256, 0, stream>>>(
      qweight, alpha, beta, in_reorder, out_reorder, block_bw, wT);

  bcq_gemm_wmma<<<dim3(BATCH / BM, OUT_F / BN), 256, 0, stream>>>(x, wT, out);
}
